// SwinUNetModel_18554258719360
// MI455X (gfx1250) — matrix operand order repaired and checked by reading, not compiled, not run on hardware
//
#include <hip/hip_runtime.h>
#include <math.h>

typedef _Float16 half_t;
typedef __attribute__((ext_vector_type(16))) _Float16 v16h;
typedef __attribute__((ext_vector_type(8)))  float    v8f;

#define DISP 4
#define ATT_SCALE 0.17677669529663687f  /* 32^-0.5 */

// ---------------------------------------------------------------------------
// WMMA helpers (CDNA5 wave32, V_WMMA_F32_16X16X32_F16)
// ---------------------------------------------------------------------------
__device__ __forceinline__ v8f wmma32(v16h a, v16h b, v8f c) {
  return __builtin_amdgcn_wmma_f32_16x16x32_f16(false, a, false, b, (short)0, c,
                                                false, false);
}

// A-matrix 16x32 f16 fragment from LDS tile (row-major, `stride` halfs/row).
// ISA layout: lane L<16 -> row L, K in {8g..8g+7} U {16+8g..16+8g+7}, g=L>>4.
__device__ __forceinline__ v16h load_afrag(const half_t* base, int lane, int stride) {
  int r = lane & 15, g = lane >> 4;
  const half_t* p = base + r * stride + g * 8;
  v16h a;
#pragma unroll
  for (int i = 0; i < 8; ++i) a[i] = p[i];
#pragma unroll
  for (int i = 0; i < 8; ++i) a[i + 8] = p[16 + i];
  return a;
}

// B-matrix 32x16 f16 fragment, sourced from B^T stored row-major in LDS
// (row = output column n, `stride` halfs/row). Lane L: col L&15, K in {8g..8g+7} U {16+8g..16+8g+7}.
__device__ __forceinline__ v16h load_bfrag(const half_t* baseT, int lane, int stride) {
  int c = lane & 15, g = lane >> 4;
  const half_t* p = baseT + c * stride + g * 8;
  v16h b;
#pragma unroll
  for (int i = 0; i < 8; ++i) b[i] = p[i];
#pragma unroll
  for (int i = 0; i < 8; ++i) b[i + 8] = p[16 + i];
  return b;
}

// ---------------------------------------------------------------------------
// fp32 -> f16 weight conversion
// ---------------------------------------------------------------------------
__global__ void k_cvt(const float* __restrict__ s, half_t* __restrict__ d, int n) {
  int i = blockIdx.x * blockDim.x + threadIdx.x;
  if (i < n) d[i] = (half_t)s[i];
}

// ---------------------------------------------------------------------------
// Patch merging: t[n,c] = gather(x)[n,0:256] @ Wm + bm   (M=131072,K=256,N=128)
// block: 128 thr (4 waves), 64x64 output tile
// ---------------------------------------------------------------------------
__global__ void k_patch_merge(const float* __restrict__ x, const half_t* __restrict__ Wm,
                              const float* __restrict__ bm, float* __restrict__ outT) {
  __shared__ __align__(16) half_t sA[64 * 32];
  __shared__ __align__(16) half_t sBT[64 * 32];
  int tid = threadIdx.x, lane = tid & 31, w = tid >> 5, g = lane >> 4;
  int m0 = blockIdx.x * 64, nc0 = blockIdx.y * 64;
  v8f z = {};
  v8f acc[4];
#pragma unroll
  for (int t = 0; t < 4; ++t) acc[t] = z;

  for (int k0 = 0; k0 < 256; k0 += 32) {
    for (int e = tid; e < 2048; e += 128) {
      int row = e >> 5, col = e & 31;
      int tok = m0 + row;
      int bi = tok >> 14, rem = tok & 16383, oh = rem >> 7, ow = rem & 127;
      int k = k0 + col;
      int cc = k >> 2, ki = (k >> 1) & 1, kj = k & 1;
      sA[e] = (half_t)x[(((size_t)bi * 64 + cc) * 256 + (oh * 2 + ki)) * 256 +
                        (ow * 2 + kj)];
    }
    for (int e = tid; e < 256; e += 128) {  // 32 j * 8 n-chunks
      int j = e & 31, n8 = (e >> 5) * 8;
      const half_t* wp = &Wm[(size_t)(k0 + j) * 128 + nc0 + n8];
#pragma unroll
      for (int i = 0; i < 8; ++i) sBT[(n8 + i) * 32 + j] = wp[i];
    }
    __syncthreads();
    v16h a = load_afrag(&sA[(w * 16) * 32], lane, 32);
#pragma unroll
    for (int nt = 0; nt < 4; ++nt) {
      v16h b = load_bfrag(&sBT[(nt * 16) * 32], lane, 32);
      acc[nt] = wmma32(a, b, acc[nt]);
    }
    __syncthreads();
  }
#pragma unroll
  for (int nt = 0; nt < 4; ++nt) {
    int col = nc0 + nt * 16 + (lane & 15);
    float bv = bm[col];
#pragma unroll
    for (int r = 0; r < 8; ++r) {
      size_t row = (size_t)m0 + w * 16 + r + 8 * g;
      outT[row * 128 + col] = acc[nt][r] + bv;
    }
  }
}

// ---------------------------------------------------------------------------
// Generic GEMM: out = A(f16)[M,K] @ W(f16)[K,N], tile 64 x (NT*16)
// EPI 0: store f16 to outH.  EPI 1: +bias +residual(tres), store fp32 in place.
// ---------------------------------------------------------------------------
template <int NT, int EPI>
__global__ void k_gemm(const half_t* __restrict__ A, const half_t* __restrict__ Wt,
                       const float* __restrict__ bias, float* __restrict__ tres,
                       half_t* __restrict__ outH, int K, int lda, int ldw, int ldo) {
  __shared__ __align__(16) half_t sA[64 * 32];
  __shared__ __align__(16) half_t sBT[NT * 16 * 32];
  int tid = threadIdx.x, lane = tid & 31, w = tid >> 5, g = lane >> 4;
  int m0 = blockIdx.x * 64, nc0 = blockIdx.y * (NT * 16);
  v8f z = {};
  v8f acc[NT];
#pragma unroll
  for (int t = 0; t < NT; ++t) acc[t] = z;

  for (int k0 = 0; k0 < K; k0 += 32) {
    for (int e8 = tid; e8 < 256; e8 += 128) {
      int row = e8 >> 2, c8 = (e8 & 3) * 8;
      *(uint4*)&sA[row * 32 + c8] =
          *(const uint4*)&A[(size_t)(m0 + row) * lda + k0 + c8];
    }
    for (int e = tid; e < 64 * NT; e += 128) {  // 32 j * (NT*2) n-chunks
      int j = e & 31, n8 = (e >> 5) * 8;
      const half_t* wp = &Wt[(size_t)(k0 + j) * ldw + nc0 + n8];
#pragma unroll
      for (int i = 0; i < 8; ++i) sBT[(n8 + i) * 32 + j] = wp[i];
    }
    __syncthreads();
    v16h a = load_afrag(&sA[(w * 16) * 32], lane, 32);
#pragma unroll
    for (int nt = 0; nt < NT; ++nt) {
      v16h b = load_bfrag(&sBT[(nt * 16) * 32], lane, 32);
      acc[nt] = wmma32(a, b, acc[nt]);
    }
    __syncthreads();
  }
#pragma unroll
  for (int nt = 0; nt < NT; ++nt) {
    int col = nc0 + nt * 16 + (lane & 15);
    float bv = 0.0f;
    if (EPI == 1) bv = bias[col];
#pragma unroll
    for (int r = 0; r < 8; ++r) {
      size_t row = (size_t)m0 + w * 16 + r + 8 * g;
      if (EPI == 0) {
        outH[row * ldo + col] = (half_t)acc[nt][r];
      } else {
        float v = acc[nt][r] + bv + tres[row * ldo + col];
        tres[row * ldo + col] = v;
      }
    }
  }
}

// ---------------------------------------------------------------------------
// LayerNorm over C=128, one wave per token, output f16
// ---------------------------------------------------------------------------
__global__ void k_ln(const float* __restrict__ t, const float* __restrict__ gam,
                     const float* __restrict__ bet, half_t* __restrict__ out) {
  int lane = threadIdx.x & 31;
  size_t tok = (size_t)blockIdx.x * 8 + (threadIdx.x >> 5);
  const float* p = t + tok * 128;
  float v0 = p[lane], v1 = p[lane + 32], v2 = p[lane + 64], v3 = p[lane + 96];
  float s = v0 + v1 + v2 + v3;
  float q = v0 * v0 + v1 * v1 + v2 * v2 + v3 * v3;
#pragma unroll
  for (int m = 16; m >= 1; m >>= 1) {
    s += __shfl_xor(s, m);
    q += __shfl_xor(q, m);
  }
  float mean = s * (1.0f / 128.0f);
  float var = q * (1.0f / 128.0f) - mean * mean;
  float rs = rsqrtf(var + 1e-5f);
  half_t* o = out + tok * 128;
  o[lane]      = (half_t)((v0 - mean) * rs * gam[lane]      + bet[lane]);
  o[lane + 32] = (half_t)((v1 - mean) * rs * gam[lane + 32] + bet[lane + 32]);
  o[lane + 64] = (half_t)((v2 - mean) * rs * gam[lane + 64] + bet[lane + 64]);
  o[lane + 96] = (half_t)((v3 - mean) * rs * gam[lane + 96] + bet[lane + 96]);
}

// ---------------------------------------------------------------------------
// Window attention: block = 96 thr (3 waves = 3 heads), one 8x8 window.
// Shift handled as an index remap (y+4)%128 on gather and scatter.
// Dyn LDS: [S: 3*4096 f32][pos: 232 f32][q/k/vT: 3*6144 f16] = 86944 B
// ---------------------------------------------------------------------------
template <bool SHIFT>
__global__ void __launch_bounds__(96)
k_attn(const half_t* __restrict__ qkv, const float* __restrict__ pos,
       half_t* __restrict__ outO) {
  extern __shared__ char smem[];
  int tid = threadIdx.x, lane = tid & 31, h = tid >> 5, g = lane >> 4;
  int bi = blockIdx.y, win = blockIdx.x, wy = win >> 4, wx = win & 15;

  float* sSall = (float*)smem;
  float* sPos = sSall + 3 * 4096;
  half_t* hbase = (half_t*)(sPos + 232);
  float* sS = sSall + h * 4096;
  half_t* sq = hbase + h * 6144;
  half_t* sk = sq + 2048;
  half_t* svT = sk + 2048;
  half_t* sP = (half_t*)sS;  // reused in place after softmax, stride 128 halfs

  for (int e = tid; e < 225; e += 96) sPos[e] = pos[e];

  // gather q,k (row-major 64x32) and v transposed (32x64)
#pragma unroll
  for (int u = 0; u < 2; ++u) {
    int t = lane + u * 32;
    int ty = t >> 3, tx = t & 7;
    int y = wy * 8 + ty, x = wx * 8 + tx;
    if (SHIFT) { y = (y + DISP) & 127; x = (x + DISP) & 127; }
    size_t row = ((size_t)bi * 128 + y) * 128 + x;
    const half_t* src = qkv + row * 288 + h * 32;
#pragma unroll
    for (int c4 = 0; c4 < 4; ++c4)
      *(uint4*)&sq[t * 32 + c4 * 8] = *(const uint4*)&src[c4 * 8];
#pragma unroll
    for (int c4 = 0; c4 < 4; ++c4)
      *(uint4*)&sk[t * 32 + c4 * 8] = *(const uint4*)&src[96 + c4 * 8];
    half_t vloc[32];
#pragma unroll
    for (int c4 = 0; c4 < 4; ++c4)
      *(uint4*)&vloc[c4 * 8] = *(const uint4*)&src[192 + c4 * 8];
#pragma unroll
    for (int d = 0; d < 32; ++d) svT[d * 64 + t] = vloc[d];
  }
  __syncthreads();

  // S = q @ k^T * scale + relpos (+ shift mask)
#pragma unroll
  for (int tr = 0; tr < 4; ++tr) {
    v16h a = load_afrag(&sq[tr * 16 * 32], lane, 32);
#pragma unroll
    for (int tc = 0; tc < 4; ++tc) {
      v16h b = load_bfrag(&sk[tc * 16 * 32], lane, 32);
      v8f z = {};
      v8f acc = wmma32(a, b, z);
      int j = tc * 16 + (lane & 15);
      int yj = j >> 3, xj = j & 7;
#pragma unroll
      for (int r = 0; r < 8; ++r) {
        int i = tr * 16 + r + 8 * g;
        int yi = i >> 3, xi = i & 7;
        // bias[i][j] = pos[(yj-yi)+7][(xj-xi)+7]
        float v = acc[r] * ATT_SCALE + sPos[(yj - yi + 7) * 15 + (xj - xi + 7)];
        if (SHIFT) {
          if (wy == 15 && ((yi >= DISP) != (yj >= DISP))) v = -1e30f;
          if (wx == 15 && ((xi >= DISP) != (xj >= DISP))) v = -1e30f;
        }
        sS[i * 64 + j] = v;
      }
    }
  }
  __syncthreads();

  // softmax rows; write normalized probs as f16 in place (stride 128)
#pragma unroll
  for (int u = 0; u < 2; ++u) {
    int i = lane * 2 + u;
    float vals[64];
#pragma unroll
    for (int j = 0; j < 64; ++j) vals[j] = sS[i * 64 + j];
    float mx = -1e30f;
#pragma unroll
    for (int j = 0; j < 64; ++j) mx = fmaxf(mx, vals[j]);
    float sum = 0.0f;
#pragma unroll
    for (int j = 0; j < 64; ++j) { vals[j] = expf(vals[j] - mx); sum += vals[j]; }
    float inv = 1.0f / sum;
#pragma unroll
    for (int j = 0; j < 64; ++j) sP[i * 128 + j] = (half_t)(vals[j] * inv);
  }
  __syncthreads();

  // O = P @ V  (K=64, N=32)
#pragma unroll
  for (int tr = 0; tr < 4; ++tr) {
    v8f z = {};
    v8f acc[2];
    acc[0] = z; acc[1] = z;
#pragma unroll
    for (int ks = 0; ks < 2; ++ks) {
      v16h a = load_afrag(&sP[tr * 16 * 128 + ks * 32], lane, 128);
#pragma unroll
      for (int nc = 0; nc < 2; ++nc) {
        v16h b = load_bfrag(&svT[nc * 16 * 64 + ks * 32], lane, 64);
        acc[nc] = wmma32(a, b, acc[nc]);
      }
    }
#pragma unroll
    for (int nc = 0; nc < 2; ++nc) {
      int d = nc * 16 + (lane & 15);
#pragma unroll
      for (int r = 0; r < 8; ++r) {
        int i = tr * 16 + r + 8 * g;
        int ty = i >> 3, tx = i & 7;
        int y = wy * 8 + ty, x = wx * 8 + tx;
        if (SHIFT) { y = (y + DISP) & 127; x = (x + DISP) & 127; }
        size_t row = ((size_t)bi * 128 + y) * 128 + x;
        outO[row * 96 + h * 32 + d] = (half_t)acc[nc][r];
      }
    }
  }
}

// ---------------------------------------------------------------------------
// Fused MLP: t += gelu(h@W1+b1) @ W2 + b2. 64 tokens/block; U (64x512 f16)
// lives entirely in LDS (320KB/WGP makes this cheap on CDNA5).
// Dyn LDS: sH 16KB + sU 64KB + sBT 8KB = 90112 B
// ---------------------------------------------------------------------------
__global__ void __launch_bounds__(128)
k_mlp(const half_t* __restrict__ Hin, const half_t* __restrict__ W1,
      const float* __restrict__ b1, const half_t* __restrict__ W2,
      const float* __restrict__ b2, float* __restrict__ tres) {
  extern __shared__ char smem[];
  half_t* sH = (half_t*)smem;   // 64*128
  half_t* sU = sH + 64 * 128;   // 64*512
  half_t* sBT = sU + 64 * 512;  // up to 128*32
  int tid = threadIdx.x, lane = tid & 31, w = tid >> 5, g = lane >> 4;
  size_t m0 = (size_t)blockIdx.x * 64;

  for (int e8 = tid; e8 < 1024; e8 += 128) {
    int row = e8 >> 4, c8 = (e8 & 15) * 8;
    *(uint4*)&sH[row * 128 + c8] = *(const uint4*)&Hin[(m0 + row) * 128 + c8];
  }
  __syncthreads();

  // FC1 + exact gelu -> sU
  for (int nb = 0; nb < 8; ++nb) {
    v8f z = {};
    v8f acc[4];
#pragma unroll
    for (int t = 0; t < 4; ++t) acc[t] = z;
    for (int k0 = 0; k0 < 128; k0 += 32) {
      for (int e = tid; e < 256; e += 128) {
        int j = e & 31, n8 = (e >> 5) * 8;
        const half_t* wp = &W1[(size_t)(k0 + j) * 512 + nb * 64 + n8];
#pragma unroll
        for (int i = 0; i < 8; ++i) sBT[(n8 + i) * 32 + j] = wp[i];
      }
      __syncthreads();
      v16h a = load_afrag(&sH[(w * 16) * 128 + k0], lane, 128);
#pragma unroll
      for (int nt = 0; nt < 4; ++nt) {
        v16h b = load_bfrag(&sBT[(nt * 16) * 32], lane, 32);
        acc[nt] = wmma32(a, b, acc[nt]);
      }
      __syncthreads();
    }
#pragma unroll
    for (int nt = 0; nt < 4; ++nt) {
      int col = nb * 64 + nt * 16 + (lane & 15);
      float bv = b1[col];
#pragma unroll
      for (int r = 0; r < 8; ++r) {
        int row = w * 16 + r + 8 * g;
        float v = acc[nt][r] + bv;
        v = 0.5f * v * (1.0f + erff(v * 0.70710678118654752f));
        sU[row * 512 + col] = (half_t)v;
      }
    }
  }
  __syncthreads();

  // FC2 + bias + residual (in place fp32)
  v8f z2 = {};
  v8f acc2[8];
#pragma unroll
  for (int t = 0; t < 8; ++t) acc2[t] = z2;
  for (int k0 = 0; k0 < 512; k0 += 32) {
    for (int e = tid; e < 512; e += 128) {
      int j = e & 31, n8 = (e >> 5) * 8;
      const half_t* wp = &W2[(size_t)(k0 + j) * 128 + n8];
#pragma unroll
      for (int i = 0; i < 8; ++i) sBT[(n8 + i) * 32 + j] = wp[i];
    }
    __syncthreads();
    v16h a = load_afrag(&sU[(w * 16) * 512 + k0], lane, 512);
#pragma unroll
    for (int nt = 0; nt < 8; ++nt) {
      v16h b = load_bfrag(&sBT[(nt * 16) * 32], lane, 32);
      acc2[nt] = wmma32(a, b, acc2[nt]);
    }
    __syncthreads();
  }
#pragma unroll
  for (int nt = 0; nt < 8; ++nt) {
    int col = nt * 16 + (lane & 15);
    float bv = b2[col];
#pragma unroll
    for (int r = 0; r < 8; ++r) {
      size_t row = m0 + w * 16 + r + 8 * g;
      float v = acc2[nt][r] + bv + tres[row * 128 + col];
      tres[row * 128 + col] = v;
    }
  }
}

// ---------------------------------------------------------------------------
// (b, H, W, C) -> (b, C, H, W), tiled 32x32
// ---------------------------------------------------------------------------
__global__ void k_transpose(const float* __restrict__ t, float* __restrict__ out) {
  __shared__ float tile[32][33];
  int bi = blockIdx.z;
  int p0 = blockIdx.x * 32, c0 = blockIdx.y * 32;
#pragma unroll
  for (int i = 0; i < 4; ++i) {
    int p = p0 + threadIdx.y + i * 8;
    tile[threadIdx.y + i * 8][threadIdx.x] =
        t[((size_t)bi * 16384 + p) * 128 + c0 + threadIdx.x];
  }
  __syncthreads();
#pragma unroll
  for (int i = 0; i < 4; ++i) {
    int c = c0 + threadIdx.y + i * 8;
    out[((size_t)bi * 128 + c) * 16384 + p0 + threadIdx.x] =
        tile[threadIdx.x][threadIdx.y + i * 8];
  }
}

// ---------------------------------------------------------------------------
extern "C" void kernel_launch(void* const* d_in, const int* in_sizes, int n_in,
                              void* d_out, int out_size, void* d_ws, size_t ws_size,
                              hipStream_t stream) {
  (void)in_sizes; (void)n_in; (void)out_size; (void)ws_size;
  const float* x  = (const float*)d_in[0];
  const float* Wm = (const float*)d_in[1];
  const float* bm = (const float*)d_in[2];

  char* ws = (char*)d_ws;
  half_t* w16 = (half_t*)ws;
  float*  bufT   = (float*)(ws + 1048576ull);
  half_t* bufLN  = (half_t*)(ws + 68157440ull);   // 1MB + 131072*128*4
  half_t* bufQKV = (half_t*)(ws + 101711872ull);  // + 131072*128*2
  half_t* bufO   = (half_t*)(ws + 177209344ull);  // + 131072*288*2

  half_t* wWm = w16;
  half_t *wQ[2], *wO[2], *w1[2], *w2[2];
  size_t off = 32768;
  for (int b = 0; b < 2; ++b) {
    wQ[b] = w16 + off; off += 36864;
    wO[b] = w16 + off; off += 12288;
    w1[b] = w16 + off; off += 65536;
    w2[b] = w16 + off; off += 65536;
  }

  k_cvt<<<(32768 + 255) / 256, 256, 0, stream>>>(Wm, wWm, 32768);
  for (int b = 0; b < 2; ++b) {
    int base = 3 + b * 12;
    k_cvt<<<(36864 + 255) / 256, 256, 0, stream>>>((const float*)d_in[base + 2], wQ[b], 36864);
    k_cvt<<<(12288 + 255) / 256, 256, 0, stream>>>((const float*)d_in[base + 4], wO[b], 12288);
    k_cvt<<<(65536 + 255) / 256, 256, 0, stream>>>((const float*)d_in[base + 8], w1[b], 65536);
    k_cvt<<<(65536 + 255) / 256, 256, 0, stream>>>((const float*)d_in[base + 10], w2[b], 65536);
  }

  k_patch_merge<<<dim3(2048, 2), 128, 0, stream>>>(x, wWm, bm, bufT);

  hipFuncSetAttribute(reinterpret_cast<const void*>(&k_mlp),
                      hipFuncAttributeMaxDynamicSharedMemorySize, 90112);
  hipFuncSetAttribute(reinterpret_cast<const void*>(&k_attn<false>),
                      hipFuncAttributeMaxDynamicSharedMemorySize, 86944);
  hipFuncSetAttribute(reinterpret_cast<const void*>(&k_attn<true>),
                      hipFuncAttributeMaxDynamicSharedMemorySize, 86944);

  for (int b = 0; b < 2; ++b) {
    int base = 3 + b * 12;
    const float* g1  = (const float*)d_in[base + 0];
    const float* b1  = (const float*)d_in[base + 1];
    const float* pos = (const float*)d_in[base + 3];
    const float* bo  = (const float*)d_in[base + 5];
    const float* g2  = (const float*)d_in[base + 6];
    const float* b2  = (const float*)d_in[base + 7];
    const float* bf1 = (const float*)d_in[base + 9];
    const float* bf2 = (const float*)d_in[base + 11];

    k_ln<<<16384, 256, 0, stream>>>(bufT, g1, b1, bufLN);
    k_gemm<6, 0><<<dim3(2048, 3), 128, 0, stream>>>(bufLN, wQ[b], nullptr, nullptr,
                                                    bufQKV, 128, 128, 288, 288);
    if (b == 0)
      k_attn<false><<<dim3(256, 8), 96, 86944, stream>>>(bufQKV, pos, bufO);
    else
      k_attn<true><<<dim3(256, 8), 96, 86944, stream>>>(bufQKV, pos, bufO);
    k_gemm<4, 1><<<dim3(2048, 2), 128, 0, stream>>>(bufO, wO[b], bo, bufT,
                                                    nullptr, 96, 96, 128, 128);
    k_ln<<<16384, 256, 0, stream>>>(bufT, g2, b2, bufLN);
    k_mlp<<<2048, 128, 90112, stream>>>(bufLN, w1[b], bf1, w2[b], bf2, bufT);
  }

  k_transpose<<<dim3(512, 4, 8), dim3(32, 8), 0, stream>>>(bufT, (float*)d_out);
}
